// CMAlign_mask_37503654428827
// MI455X (gfx1250) — matrix operand order repaired and checked by reading, not compiled, not run on hardware
//
#include <hip/hip_runtime.h>

// CMAlign triplet loss for MI455X (gfx1250, wave32, WMMA).
// Batched GEMMs (sim, warp) run on v_wmma_f32_16x16x32_f16 over a zero-padded
// f16 copy of feat so all hot-loop memory ops are unconditional 16B vector ops.
// prS staging uses GLOBAL_LOAD_ASYNC_TO_LDS (ASYNCcnt) when available.

typedef __attribute__((ext_vector_type(16))) _Float16 v16h;
typedef __attribute__((ext_vector_type(8)))  _Float16 v8h;
typedef __attribute__((ext_vector_type(8)))  float    v8f;
typedef __attribute__((ext_vector_type(4)))  int      v4i;
typedef __attribute__((address_space(1)))    v4i      gv4i;   // global int4
typedef __attribute__((address_space(3)))    v4i      lv4i;   // LDS int4

#define N2    64          // 2N samples
#define FD    2048        // feature dim
#define HW    162         // 18*9 positions
#define HWP   176         // positions padded to 11*16 (M/N tiles)
#define KP    192         // k padded to 6*32 for warp GEMM
#define TEMP  50.0f

__device__ __forceinline__ const float* feat_base(const float* fv, const float* ft, int b) {
    return (b < 32) ? (fv + (size_t)b * FD * HW)
                    : (ft + (size_t)(b - 32) * FD * HW);
}

__device__ __forceinline__ v8f wmma_f16(v16h a, v16h b, v8f c) {
    return __builtin_amdgcn_wmma_f32_16x16x32_f16(
        /*neg_a=*/false, a, /*neg_b=*/false, b,
        /*c_mod=*/(short)0, c, /*reuse_a=*/false, /*reuse_b=*/false);
}

__device__ __forceinline__ v16h cat8(v8h lo, v8h hi) {
    return __builtin_shufflevector(lo, hi, 0,1,2,3,4,5,6,7,8,9,10,11,12,13,14,15);
}

// ---------------------------------------------------------------------------
// fpad[b][f][0..191] = f16(feat[b][f][c]) for c<162, else 0   (one-shot pass)
// ---------------------------------------------------------------------------
__global__ __launch_bounds__(192)
void convert_kernel(const float* __restrict__ fv, const float* __restrict__ ft,
                    _Float16* __restrict__ fpad) {
    const int row = blockIdx.x;              // b*FD + f
    const int b = row >> 11, f = row & (FD - 1);
    const int c = threadIdx.x;               // 0..191
    const float* fb = feat_base(fv, ft, b);
    float v = 0.f;
    if (c < HW) v = fb[(size_t)f * HW + c];
    fpad[(size_t)row * KP + c] = (_Float16)v;
}

// ---------------------------------------------------------------------------
// mask[n][p] = minmax-normalized (text[n] . maskW[p]) ; padded cols -> 0
// ---------------------------------------------------------------------------
__global__ __launch_bounds__(256)
void mask_kernel(const float* __restrict__ text, const float* __restrict__ maskW,
                 float* __restrict__ mask) {
    const int n = blockIdx.x, t = threadIdx.x;
    __shared__ float tl[512];
    __shared__ float red[256];
    for (int i = t; i < 512; i += 256) tl[i] = text[(size_t)n * 512 + i];
    __syncthreads();
    float v = 0.f;
    if (t < HW) {
        const float* wr = maskW + (size_t)t * 512;
        float s = 0.f;
        for (int j = 0; j < 512; ++j) s += tl[j] * wr[j];
        v = s;
    }
    red[t] = (t < HW) ? v : 3.4e38f; __syncthreads();
    for (int s2 = 128; s2 > 0; s2 >>= 1) { if (t < s2) red[t] = fminf(red[t], red[t + s2]); __syncthreads(); }
    const float mn = red[0]; __syncthreads();
    red[t] = (t < HW) ? v : -3.4e38f; __syncthreads();
    for (int s2 = 128; s2 > 0; s2 >>= 1) { if (t < s2) red[t] = fmaxf(red[t], red[t + s2]); __syncthreads(); }
    const float mx = red[0]; __syncthreads();
    if (t < HWP) mask[n * HWP + t] = (t < HW) ? (v - mn) / ((mx - mn) + 1e-12f) : 0.f;
}

// ---------------------------------------------------------------------------
// rn[b][p] = 1 / max(||feat[b,:,p]||_2, 1e-12) ; padded cols -> 0
// ---------------------------------------------------------------------------
__global__ __launch_bounds__(256)
void norm_kernel(const float* __restrict__ fv, const float* __restrict__ ft,
                 float* __restrict__ rn) {
    const int b = blockIdx.x, t = threadIdx.x;
    const float* fb = feat_base(fv, ft, b);
    if (t >= HWP) return;
    if (t < HW) {
        float s = 0.f;
        for (int f = 0; f < FD; ++f) { float x = fb[(size_t)f * HW + t]; s += x * x; }
        rn[b * HWP + t] = 1.f / fmaxf(sqrtf(s), 1e-12f);
    } else {
        rn[b * HWP + t] = 0.f;
    }
}

// ---------------------------------------------------------------------------
// sim + softmax -> pr (f16, [64][176][192], zero padded)
// block = (q-tile mt, sample b); 352 threads = 11 waves, wave w owns n-tile w.
// Double-buffered LDS staging (transposed), one barrier per 32-deep k-step.
// Staging index math is pure shifts/masks (352 = 32 kk x 11 col-groups).
// ---------------------------------------------------------------------------
__global__ __launch_bounds__(352)
void sim_kernel(const _Float16* __restrict__ fpad, const float* __restrict__ rn,
                const int* __restrict__ idx, _Float16* __restrict__ pr) {
    const int b  = blockIdx.y;
    const int mt = blockIdx.x;
    const int q0 = mt * 16;
    const int pb = idx[b];
    const _Float16* fpb = fpad + (size_t)b  * FD * KP;
    const _Float16* fpk = fpad + (size_t)pb * FD * KP;

    __shared__ alignas(16) _Float16 As[2][16][40];    // A slab, transposed [m][kk]
    __shared__ alignas(16) _Float16 Bs[2][HWP][40];   // B slab, transposed [n][kk]
    __shared__ float S[16][180];                      // logits for softmax

    const int tid  = threadIdx.x;
    const int lane = tid & 31;
    const int wave = tid >> 5;
    const int ln   = lane & 15;
    const int g    = lane >> 4;
    const int n0   = wave * 16;

    // Per-thread staging coordinates: kk = tid&31, col-group cc = tid>>5 (0..10).
    const int skk = tid & 31;
    const int scc = tid >> 5;

    auto stage = [&](int f0, int buf) {
        // A slab: 32 kk x 16 m, as 64 v8h chunks (m-contiguous in global)
        if (tid < 64) {
            int kk = tid >> 1, m0 = (tid & 1) * 8;
            v8h v = *(const v8h*)(fpb + (size_t)(f0 + kk) * KP + q0 + m0);
#pragma unroll
            for (int j = 0; j < 8; ++j) As[buf][m0 + j][kk] = v[j];
        }
        // B slab: each thread copies 16 cols (32B) of row f0+skk, transposed.
        {
            const _Float16* src = fpk + (size_t)(f0 + skk) * KP + scc * 16;
            v8h v0 = *(const v8h*)src;
            v8h v1 = *(const v8h*)(src + 8);
#pragma unroll
            for (int j = 0; j < 8; ++j) Bs[buf][scc * 16 + j][skk] = v0[j];
#pragma unroll
            for (int j = 0; j < 8; ++j) Bs[buf][scc * 16 + 8 + j][skk] = v1[j];
        }
    };

    v8f acc = {};
    stage(0, 0);
    __syncthreads();
    int buf = 0;
    for (int f0 = 0; f0 < FD; f0 += 32, buf ^= 1) {
        if (f0 + 32 < FD) stage(f0 + 32, buf ^ 1);
        // A fragment: lane ln holds row M=ln; elem j -> K=(j%8)+(j/8)*16+g*8
        v8h a0 = *(const v8h*)&As[buf][ln][g * 8];
        v8h a1 = *(const v8h*)&As[buf][ln][16 + g * 8];
        // B fragment: lane holds col N=ln; elem j -> K=(j%8)+(j/8)*16+g*8
        v8h b0 = *(const v8h*)&Bs[buf][n0 + ln][g * 8];
        v8h b1 = *(const v8h*)&Bs[buf][n0 + ln][16 + g * 8];
        acc = wmma_f16(cat8(a0, a1), cat8(b0, b1), acc);
        __syncthreads();
    }

    // Scale by TEMP * rn_q * rn_k, spill logits to LDS for row softmax.
    const int kcol = n0 + ln;
    const float rk = rn[pb * HWP + kcol];
#pragma unroll
    for (int r = 0; r < 8; ++r) {
        const int mr = r + 8 * g;
        const float rq = rn[b * HWP + q0 + mr];
        S[mr][kcol] = TEMP * rq * rk * acc[r];
    }
    __syncthreads();

    for (int row = wave; row < 16; row += 11) {
        const int qg = q0 + row;
        _Float16* prow = pr + ((size_t)b * HWP + qg) * KP;
        if (qg >= HW) {
            for (int c = lane; c < KP; c += 32) prow[c] = (_Float16)0.f;
            continue;
        }
        float mx = -3.4e38f;
        for (int c = lane; c < HW; c += 32) mx = fmaxf(mx, S[row][c]);
        for (int o = 16; o > 0; o >>= 1) mx = fmaxf(mx, __shfl_xor(mx, o, 32));
        float sum = 0.f;
        for (int c = lane; c < HW; c += 32) {
            float e = __expf(S[row][c] - mx);
            S[row][c] = e;
            sum += e;
        }
        for (int o = 16; o > 0; o >>= 1) sum += __shfl_xor(sum, o, 32);
        const float inv = 1.f / sum;
        for (int c = lane; c < HW; c += 32) prow[c] = (_Float16)(S[row][c] * inv);
        for (int c = HW + lane; c < KP; c += 32) prow[c] = (_Float16)0.f;
    }
}

// ---------------------------------------------------------------------------
// warp GEMM + fused recon/distance epilogue.
// block = (q-tile, b); 256 threads = 8 waves over 128 f-tiles.
// warp[f,q] = sum_k kf[f,k] * pr[q,k]; dsum[b,q] = sum_f (mask*(x-w)+eps)^2.
// A fragments: two aligned global_load_b128 per lane per k-step from fpad.
// prS staged with async global->LDS copies when the toolchain exposes them.
// ---------------------------------------------------------------------------
template <bool STORE_RECON>
__global__ __launch_bounds__(256)
void warp_kernel(const float* __restrict__ fv, const float* __restrict__ ft,
                 const _Float16* __restrict__ fpad, const float* __restrict__ mask,
                 const int* __restrict__ idx, const _Float16* __restrict__ pr,
                 float* __restrict__ recon, float* __restrict__ dsum) {
    const int b  = blockIdx.y;
    const int qt = blockIdx.x;
    const int q0 = qt * 16;
    const int pb = idx[b];
    const float* fb = feat_base(fv, ft, b);
    const _Float16* fpk = fpad + (size_t)pb * FD * KP;

    __shared__ alignas(16) _Float16 prS[16][KP];
    __shared__ float maskS[16];
    __shared__ float dl[16];

    const int tid = threadIdx.x, lane = tid & 31, wave = tid >> 5;
    const int ln = lane & 15, g = lane >> 4;

    // Stage 16 pr rows (one contiguous 6KB block).
    const _Float16* prBase = pr + ((size_t)b * HWP + q0) * KP;
#if __has_builtin(__builtin_amdgcn_global_load_async_to_lds_b128)
    for (int i = tid; i < (16 * KP) / 8; i += 256) {
        __builtin_amdgcn_global_load_async_to_lds_b128(
            (gv4i*)(uintptr_t)(prBase + i * 8),
            (lv4i*)(uint32_t)(uintptr_t)(&prS[0][0] + i * 8),
            0, 0);
    }
#if __has_builtin(__builtin_amdgcn_s_wait_asynccnt)
    __builtin_amdgcn_s_wait_asynccnt(0);
#else
    asm volatile("s_wait_asynccnt 0x0" ::: "memory");
#endif
#else
    for (int i = tid; i < (16 * KP) / 8; i += 256) {
        *(v8h*)(&prS[0][0] + i * 8) = *(const v8h*)(prBase + i * 8);
    }
#endif
    if (tid < 16) {
        int q = q0 + tid;
        maskS[tid] = (q < HW) ? mask[b * HWP + q] : 0.f;
        dl[tid] = 0.f;
    }
    __syncthreads();

    float dtot = 0.f;
    for (int ftile = wave; ftile < FD / 16; ftile += 8) {
        const int f0 = ftile * 16;
        const _Float16* arow = fpk + (size_t)(f0 + ln) * KP;   // A row M = f0+ln
        if (ftile + 8 < FD / 16)
            __builtin_prefetch(fpk + (size_t)(f0 + 128 + ln) * KP, 0, 1);
        v8f acc = {};
#pragma unroll
        for (int k0 = 0; k0 < KP; k0 += 32) {
            v8h a0 = *(const v8h*)&arow[k0 + g * 8];
            v8h a1 = *(const v8h*)&arow[k0 + 16 + g * 8];
            v8h b0 = *(const v8h*)&prS[ln][k0 + g * 8];
            v8h b1 = *(const v8h*)&prS[ln][k0 + 16 + g * 8];
            acc = wmma_f16(cat8(a0, a1), cat8(b0, b1), acc);
        }
        const int q = q0 + ln;
        if (q < HW) {
            const float m = maskS[ln];
#pragma unroll
            for (int r = 0; r < 8; ++r) {
                const int f = f0 + r + 8 * g;
                const float x = fb[(size_t)f * HW + q];
                const float w = acc[r];
                if (STORE_RECON)
                    recon[((size_t)b * FD + f) * HW + q] = m * w + (1.f - m) * x;
                const float d = m * (x - w) + 1e-6f;
                dtot += d * d;
            }
        }
    }
    atomicAdd(&dl[ln], dtot);
    __syncthreads();
    if (tid < 16 && q0 + tid < HW) dsum[b * HW + q0 + tid] = dl[tid];
}

// ---------------------------------------------------------------------------
// comask + triplet loss partial sums
// ---------------------------------------------------------------------------
__global__ __launch_bounds__(256)
void loss_kernel(const float* __restrict__ mask, const _Float16* __restrict__ pr_pos,
                 const int* __restrict__ pos_idx, const float* __restrict__ dpos,
                 const float* __restrict__ dneg, float* __restrict__ lossAcc) {
    const int b = blockIdx.x, t = threadIdx.x;
    const int pb = pos_idx[b];
    __shared__ float red[256];
    float cm = 0.f;
    if (t < HW) {
        const _Float16* prow = pr_pos + ((size_t)b * HWP + t) * KP;
        float s = 0.f;
        for (int k = 0; k < HW; ++k) s += (float)prow[k] * mask[pb * HWP + k];
        cm = mask[b * HWP + t] * s;
    }
    red[t] = (t < HW) ? cm : 3.4e38f; __syncthreads();
    for (int s2 = 128; s2 > 0; s2 >>= 1) { if (t < s2) red[t] = fminf(red[t], red[t + s2]); __syncthreads(); }
    const float mn = red[0]; __syncthreads();
    red[t] = (t < HW) ? cm : -3.4e38f; __syncthreads();
    for (int s2 = 128; s2 > 0; s2 >>= 1) { if (t < s2) red[t] = fmaxf(red[t], red[t + s2]); __syncthreads(); }
    const float mx = red[0]; __syncthreads();
    float contrib = 0.f;
    if (t < HW) {
        const float c  = (cm - mn) / ((mx - mn) + 1e-12f);
        const float dp = sqrtf(dpos[b * HW + t]);
        const float dn = sqrtf(dneg[b * HW + t]);
        contrib = c * fmaxf(dp - dn + 0.3f, 0.f);
    }
    red[t] = contrib; __syncthreads();
    for (int s2 = 128; s2 > 0; s2 >>= 1) { if (t < s2) red[t] += red[t + s2]; __syncthreads(); }
    if (t == 0) atomicAdd(lossAcc, red[0]);
}

__global__ void zero_kernel(float* p) { *p = 0.f; }

__global__ void finalize_kernel(const float* __restrict__ lossAcc, float* __restrict__ out) {
    out[(size_t)N2 * FD * HW] = *lossAcc / (float)(N2 * HW);
}

// ---------------------------------------------------------------------------
extern "C" void kernel_launch(void* const* d_in, const int* in_sizes, int n_in,
                              void* d_out, int out_size, void* d_ws, size_t ws_size,
                              hipStream_t stream) {
    (void)in_sizes; (void)n_in; (void)out_size; (void)ws_size;
    const float* feat_v = (const float*)d_in[0];
    const float* feat_t = (const float*)d_in[1];
    const float* text   = (const float*)d_in[2];
    const float* maskW  = (const float*)d_in[3];
    const int*   posIdx = (const int*)d_in[4];
    const int*   negIdx = (const int*)d_in[5];
    float* out = (float*)d_out;

    char* w = (char*)d_ws;
    float* mask = (float*)w;    w += (size_t)N2 * HWP * 4;   // 45056
    float* rn   = (float*)w;    w += (size_t)N2 * HWP * 4;   // 45056
    float* dpos = (float*)w;    w += (size_t)N2 * HW * 4;    // 41472
    float* dneg = (float*)w;    w += (size_t)N2 * HW * 4;    // 41472
    float* lossAcc = (float*)w; w += 256;
    _Float16* fpad   = (_Float16*)w; w += (size_t)N2 * FD * KP * 2;   // 50.3 MB
    _Float16* pr_pos = (_Float16*)w; w += (size_t)N2 * HWP * KP * 2;  // 4.3 MB
    _Float16* pr_neg = (_Float16*)w;

    convert_kernel<<<N2 * FD, 192, 0, stream>>>(feat_v, feat_t, fpad);
    mask_kernel<<<N2, 256, 0, stream>>>(text, maskW, mask);
    norm_kernel<<<N2, 256, 0, stream>>>(feat_v, feat_t, rn);

    sim_kernel<<<dim3(11, N2), 352, 0, stream>>>(fpad, rn, posIdx, pr_pos);
    sim_kernel<<<dim3(11, N2), 352, 0, stream>>>(fpad, rn, negIdx, pr_neg);

    warp_kernel<true ><<<dim3(11, N2), 256, 0, stream>>>(feat_v, feat_t, fpad, mask, posIdx, pr_pos, out, dpos);
    warp_kernel<false><<<dim3(11, N2), 256, 0, stream>>>(feat_v, feat_t, fpad, mask, negIdx, pr_neg, out, dneg);

    zero_kernel<<<1, 1, 0, stream>>>(lossAcc);
    loss_kernel<<<N2, 256, 0, stream>>>(mask, pr_pos, posIdx, dpos, dneg, lossAcc);
    finalize_kernel<<<1, 1, 0, stream>>>(lossAcc, out);
}
